// FlashAttention_73959336837540
// MI455X (gfx1250) — matrix operand order repaired and checked by reading, not compiled, not run on hardware
//
#include <hip/hip_runtime.h>
#include <hip/hip_bf16.h>
#include <math.h>

// ---- problem constants (Falcon-7B style MQA) ----
#define Bq   2
#define Sq   1024
#define Hq   4544
#define NHq  71
#define Dq   64
#define QKVO (Hq + 2 * Dq)     // 4672
#define Mq   (Bq * Sq)         // 2048

typedef __attribute__((ext_vector_type(16))) __bf16 v16bf;
typedef __attribute__((ext_vector_type(8)))  __bf16 v8bf;
typedef __attribute__((ext_vector_type(2)))  __bf16 v2bf;
typedef __attribute__((ext_vector_type(8)))  float  v8f;
typedef __attribute__((ext_vector_type(4)))  float  v4f;
typedef __attribute__((ext_vector_type(4)))  int    v4i;

// ---- CDNA5 async global->LDS copy (ASYNCcnt path), guarded ----
#if defined(__AMDGCN__) && __has_builtin(__builtin_amdgcn_global_load_async_to_lds_b128)
#define ASYNC_LDS 1
typedef __attribute__((address_space(1))) v4i gl_v4i;
typedef __attribute__((address_space(3))) v4i lds_v4i;

__device__ __forceinline__ void async_cp_b128(const __bf16* g, __bf16* l) {
  // generic LDS pointer low 32 bits == LDS byte offset (aperture rule, ISA 10.2)
  __builtin_amdgcn_global_load_async_to_lds_b128(
      (gl_v4i*)(uintptr_t)g,
      (lds_v4i*)(uint32_t)(uintptr_t)l, 0, 0);
}
__device__ __forceinline__ void wait_async0() {
#if __has_builtin(__builtin_amdgcn_s_wait_asynccnt)
  __builtin_amdgcn_s_wait_asynccnt(0);
#else
  asm volatile("s_wait_asynccnt 0x0" ::: "memory");
#endif
}
#endif

__device__ __forceinline__ v16bf mk16(v8bf lo, v8bf hi) {
  v16bf r;
#pragma unroll
  for (int i = 0; i < 8; ++i) { r[i] = lo[i]; r[i + 8] = hi[i]; }
  return r;
}

// ---------------- f32 -> bf16 convert (8 elems/thread, b128 traffic) ----------------
__global__ void f2bf_kernel(const float* __restrict__ src,
                            __bf16* __restrict__ dst, size_t n) {
  size_t i = ((size_t)blockIdx.x * blockDim.x + threadIdx.x) * 8;
  if (i + 8 <= n) {
    v4f a = *(const v4f*)(src + i);
    v4f b = *(const v4f*)(src + i + 4);
    v8bf o;
#pragma unroll
    for (int j = 0; j < 4; ++j) { o[j] = (__bf16)a[j]; o[4 + j] = (__bf16)b[j]; }
    *(v8bf*)(dst + i) = o;
  } else {
    for (; i < n; ++i) dst[i] = (__bf16)src[i];
  }
}

// ---------------- WMMA GEMM: C[M][N] = A[M][K] @ W[K][N] + bias ----------------
// block tile 64x64, K-step 32, 128 threads = 4 waves; wave w -> rows [16w,16w+16)
__global__ __launch_bounds__(128) void gemm_bf16_kernel(
    const __bf16* __restrict__ A, const __bf16* __restrict__ W,
    const float* __restrict__ bias, float* __restrict__ C,
    int M, int N, int K) {
  __shared__ __bf16 As[64][32];  // [m][k]
  __shared__ __bf16 Bs[64][32];  // [n][k]  (transposed tile)

  const int tid  = threadIdx.x;
  const int wave = tid >> 5;
  const int lane = tid & 31;
  const int bm = blockIdx.y * 64;
  const int bn = blockIdx.x * 64;

  v8f acc[4];
#pragma unroll
  for (int nt = 0; nt < 4; ++nt)
#pragma unroll
    for (int i = 0; i < 8; ++i) acc[nt][i] = 0.f;

  const int ksel = (lane >> 4) * 8;     // A frag K-chunk base
  const int kbB  = (lane >> 4) * 16;    // unused; B frags also use ksel
  const int rowA = (wave << 4) + (lane & 15);

  // W-transpose thread mapping: k-row pair + 8 columns
  const int kp  = tid >> 3;             // 0..15 -> k rows 2kp, 2kp+1
  const int ng8 = (tid & 7) * 8;        // 8 columns

  for (int k0 = 0; k0 < K; k0 += 32) {
    { // A tile: 64x32, straight copy -> async to LDS when available
      int r = tid >> 1, s = (tid & 1) * 16;
      const __bf16* src = A + (size_t)(bm + r) * K + k0 + s;
#if defined(ASYNC_LDS)
      async_cp_b128(src,     &As[r][s]);
      async_cp_b128(src + 8, &As[r][s + 8]);
#else
      *(v8bf*)&As[r][s]     = *(const v8bf*)src;
      *(v8bf*)&As[r][s + 8] = *(const v8bf*)(src + 8);
#endif
    }
    { // W tile 32x64 -> Bs[n][k] transposed; pack k-pairs -> b32 LDS stores
      const __bf16* src0 = W + (size_t)(k0 + 2 * kp) * N + bn + ng8;
      v8bf wa = *(const v8bf*)src0;
      v8bf wb = *(const v8bf*)(src0 + N);
#pragma unroll
      for (int j = 0; j < 8; ++j) {
        v2bf p; p[0] = wa[j]; p[1] = wb[j];
        *(v2bf*)&Bs[ng8 + j][2 * kp] = p;
      }
    }
    if (k0 + 32 < K) { // near-cache prefetch of next tiles
      __builtin_prefetch(A + (size_t)(bm + (tid >> 1)) * K + k0 + 32, 0, 3);
      __builtin_prefetch(W + (size_t)(k0 + 32 + 2 * kp) * N + bn + ng8, 0, 3);
    }
#if defined(ASYNC_LDS)
    wait_async0();
#endif
    __syncthreads();

    // gather all fragments first, then issue WMMAs back-to-back
    v16bf af = mk16(*(const v8bf*)&As[rowA][ksel],
                    *(const v8bf*)&As[rowA][ksel + 16]);
    v16bf bfr[4];
#pragma unroll
    for (int nt = 0; nt < 4; ++nt) {
      int col = nt * 16 + (lane & 15);
      bfr[nt] = mk16(*(const v8bf*)&Bs[col][ksel],
                     *(const v8bf*)&Bs[col][ksel + 16]);
    }
#pragma unroll
    for (int nt = 0; nt < 4; ++nt)
      acc[nt] = __builtin_amdgcn_wmma_f32_16x16x32_bf16(
          false, af, false, bfr[nt], (short)0, acc[nt], false, false);
    __syncthreads();
  }

  const int lh8 = (lane >> 4) << 3;
  const int ln  = lane & 15;
#pragma unroll
  for (int nt = 0; nt < 4; ++nt) {
    int col = bn + nt * 16 + ln;
    float bs = bias[col];
#pragma unroll
    for (int r = 0; r < 8; ++r) {
      int row = bm + (wave << 4) + lh8 + r;
      C[(size_t)row * N + col] = acc[nt][r] + bs;
    }
  }
}

// ---------------- RoPE + split fused -> q/k/v (bf16) ----------------
// fused: [B][S][NH+2][D] f32 ; q: [B][NH][S][D] ; k,v: [B][S][D]
__global__ void rope_split_kernel(const float* __restrict__ fused,
                                  __bf16* __restrict__ q,
                                  __bf16* __restrict__ k,
                                  __bf16* __restrict__ v) {
  size_t idx = (size_t)blockIdx.x * blockDim.x + threadIdx.x;
  const size_t total = (size_t)Mq * QKVO;
  if (idx >= total) return;
  int d = (int)(idx & 63);
  size_t t = idx >> 6;                 // D = 64
  int h = (int)(t % (NHq + 2));
  size_t bs = t / (NHq + 2);
  int s = (int)(bs % Sq);
  int b = (int)(bs / Sq);

  float x = fused[idx];
  if (h == NHq + 1) { v[(bs << 6) + d] = (__bf16)x; return; }

  // inv_freq = 10000^(-(d&31)/32) = exp(-(d&31) * ln(10000)/32)
  float ang = (float)s * __expf(-0.28782313662425572f * (float)(d & 31));
  float c = __cosf(ang), sn = __sinf(ang);
  float partner = fused[idx ^ 32];
  float sign = (d < 32) ? -1.f : 1.f;
  float val = x * c + sign * partner * sn;

  if (h == NHq) k[(bs << 6) + d] = (__bf16)val;
  else q[((((size_t)b * NHq + h) * Sq) + s) * Dq + d] = (__bf16)val;
}

// ---------------- flash attention (MQA, causal) ----------------
// grid.x = B*NH, grid.y = S/64 ; 128 threads = 4 waves, wave -> 16 query rows
__global__ __launch_bounds__(128) void mqa_flash_kernel(
    const __bf16* __restrict__ Q,   // [B][NH][S][D]
    const __bf16* __restrict__ Kk,  // [B][S][D]
    const __bf16* __restrict__ Vv,  // [B][S][D]
    __bf16* __restrict__ ctx) {     // [B][S][NH*D]
  __shared__ __bf16 Ks[32][64];      // [keyLocal][d]
  __shared__ __bf16 Vt[64][32];      // [d][keyLocal]
  __shared__ float  Ssc[4][16][32];  // per-wave raw scores
  __shared__ __bf16 Ps[4][16][32];   // per-wave probs (bf16)
  __shared__ float  Fac[4][16];
  __shared__ float  Linv[4][16];

  const int bh = blockIdx.x;
  const int b = bh / NHq, h = bh % NHq;
  const int qbase = blockIdx.y * 64;
  const int tid = threadIdx.x, wave = tid >> 5, lane = tid & 31;

  const __bf16* Qp = Q + ((size_t)b * NHq + h) * Sq * Dq;
  const __bf16* Kp = Kk + (size_t)b * Sq * Dq;
  const __bf16* Vp = Vv + (size_t)b * Sq * Dq;

  // Q fragments (rows fixed for whole kernel)
  const int qrow = qbase + (wave << 4) + (lane & 15);
  const int ksel = (lane >> 4) * 8;
  v16bf qf[2];
#pragma unroll
  for (int kh = 0; kh < 2; ++kh) {
    const __bf16* src = Qp + (size_t)qrow * Dq + kh * 32 + ksel;
    qf[kh] = mk16(*(const v8bf*)src, *(const v8bf*)(src + 16));
  }

  v8f oacc[4];
#pragma unroll
  for (int nt = 0; nt < 4; ++nt)
#pragma unroll
    for (int i = 0; i < 8; ++i) oacc[nt][i] = 0.f;

  float m_i = -__builtin_inff(), l_i = 0.f;   // valid on lanes 0..15 (row = lane)
  const int lh8 = (lane >> 4) << 3;
  const int ln  = lane & 15;
  const int kp  = tid >> 3;           // V-transpose: key pair 2kp, 2kp+1
  const int dg8 = (tid & 7) * 8;      // 8 d-values
  const int nkt = qbase / 32 + 2;     // covers keys <= qbase+63

  for (int kt = 0; kt < nkt; ++kt) {
    const int kb = kt * 32;
    { // cooperative K/V tile load; K straight copy (async), V transposed (sync)
      int key = tid >> 2, dg = (tid & 3) * 16;
      const __bf16* ksrc = Kp + (size_t)(kb + key) * Dq + dg;
#if defined(ASYNC_LDS)
      async_cp_b128(ksrc,     &Ks[key][dg]);
      async_cp_b128(ksrc + 8, &Ks[key][dg + 8]);
#else
      *(v8bf*)&Ks[key][dg]     = *(const v8bf*)ksrc;
      *(v8bf*)&Ks[key][dg + 8] = *(const v8bf*)(ksrc + 8);
#endif
      const __bf16* vsrc = Vp + (size_t)(kb + 2 * kp) * Dq + dg8;
      v8bf va = *(const v8bf*)vsrc;
      v8bf vb = *(const v8bf*)(vsrc + Dq);
#pragma unroll
      for (int j = 0; j < 8; ++j) {
        v2bf p; p[0] = va[j]; p[1] = vb[j];
        *(v2bf*)&Vt[dg8 + j][2 * kp] = p;
      }
      if (kt + 1 < nkt) {
        __builtin_prefetch(Kp + (size_t)(kb + 32 + key) * Dq + dg, 0, 3);
        __builtin_prefetch(Vp + (size_t)(kb + 32 + 2 * kp) * Dq + dg8, 0, 3);
      }
    }
#if defined(ASYNC_LDS)
    wait_async0();
#endif
    __syncthreads();

    // scores: S[16][32] = Q(16x64) @ K^T(64x32), two 16-col panels
#pragma unroll
    for (int n16 = 0; n16 < 2; ++n16) {
      int colKeyLocal = n16 * 16 + ln;
      v16bf kf[2];
#pragma unroll
      for (int kh = 0; kh < 2; ++kh) {
        int dbase = kh * 32 + (lane >> 4) * 8;
        kf[kh] = mk16(*(const v8bf*)&Ks[colKeyLocal][dbase],
                      *(const v8bf*)&Ks[colKeyLocal][dbase + 16]);
      }
      v8f s;
#pragma unroll
      for (int i = 0; i < 8; ++i) s[i] = 0.f;
      s = __builtin_amdgcn_wmma_f32_16x16x32_bf16(
          false, qf[0], false, kf[0], (short)0, s, false, false);
      s = __builtin_amdgcn_wmma_f32_16x16x32_bf16(
          false, qf[1], false, kf[1], (short)0, s, false, false);

      int colKey = kb + colKeyLocal;
#pragma unroll
      for (int r = 0; r < 8; ++r) {
        int row = qbase + (wave << 4) + lh8 + r;
        float val = s[r] * 0.125f;                      // 1/sqrt(64)
        if (colKey > row) val = -__builtin_inff();      // causal
        Ssc[wave][lh8 + r][colKeyLocal] = val;
      }
    }

    // online softmax: lane < 16 owns row = lane (wave-local LDS, DS in-order)
    if (lane < 16) {
      int row = lane;
      float mx = -__builtin_inff();
#pragma unroll
      for (int j = 0; j < 32; ++j) mx = fmaxf(mx, Ssc[wave][row][j]);
      float mnew = fmaxf(m_i, mx);
      float fac = __expf(m_i - mnew);
      float rs = 0.f;
#pragma unroll
      for (int j = 0; j < 32; ++j) {
        float p = __expf(Ssc[wave][row][j] - mnew);
        rs += p;
        Ps[wave][row][j] = (__bf16)p;
      }
      l_i = l_i * fac + rs;
      m_i = mnew;
      Fac[wave][row] = fac;
    }

    // rescale running output by row factors
    float fr[8];
#pragma unroll
    for (int r = 0; r < 8; ++r) fr[r] = Fac[wave][lh8 + r];
#pragma unroll
    for (int nt = 0; nt < 4; ++nt)
#pragma unroll
      for (int r = 0; r < 8; ++r) oacc[nt][r] *= fr[r];

    // P fragment (A: 16x32); gather all V fragments, then 4 WMMAs back-to-back
    v16bf pf = mk16(*(const v8bf*)&Ps[wave][ln][ksel],
                    *(const v8bf*)&Ps[wave][ln][ksel + 16]);
    v16bf vfr[4];
#pragma unroll
    for (int nt = 0; nt < 4; ++nt) {
      int d = nt * 16 + ln;
      int kb2 = (lane >> 4) * 8;
      vfr[nt] = mk16(*(const v8bf*)&Vt[d][kb2],
                     *(const v8bf*)&Vt[d][kb2 + 16]);
    }
#pragma unroll
    for (int nt = 0; nt < 4; ++nt)
      oacc[nt] = __builtin_amdgcn_wmma_f32_16x16x32_bf16(
          false, pf, false, vfr[nt], (short)0, oacc[nt], false, false);
    __syncthreads();
  }

  if (lane < 16) Linv[wave][lane] = 1.f / l_i;
  float li[8];
#pragma unroll
  for (int r = 0; r < 8; ++r) li[r] = Linv[wave][lh8 + r];
#pragma unroll
  for (int nt = 0; nt < 4; ++nt) {
    int col = nt * 16 + ln;
#pragma unroll
    for (int r = 0; r < 8; ++r) {
      int row = qbase + (wave << 4) + lh8 + r;
      ctx[((size_t)b * Sq + row) * (NHq * Dq) + h * Dq + col] =
          (__bf16)(oacc[nt][r] * li[r]);
    }
  }
}

// ---------------- launcher ----------------
extern "C" void kernel_launch(void* const* d_in, const int* in_sizes, int n_in,
                              void* d_out, int out_size, void* d_ws, size_t ws_size,
                              hipStream_t stream) {
  (void)in_sizes; (void)n_in; (void)out_size; (void)ws_size;
  // inputs: 0 key, 1 value, 2 query, 3 mask, 4 W_qkv, 5 b_qkv, 6 W_dense, 7 b_dense
  const float* query   = (const float*)d_in[2];
  const float* W_qkv   = (const float*)d_in[4];
  const float* b_qkv   = (const float*)d_in[5];
  const float* W_dense = (const float*)d_in[6];
  const float* b_dense = (const float*)d_in[7];
  float* out = (float*)d_out;

  const size_t nWqkv = (size_t)Hq * QKVO;
  const size_t nWd   = (size_t)Hq * Hq;
  const size_t nX    = (size_t)Mq * Hq;
  const size_t nQ    = (size_t)Bq * NHq * Sq * Dq;
  const size_t nKV   = (size_t)Bq * Sq * Dq;
  const size_t nCtx  = (size_t)Mq * Hq;
  const size_t nFused= (size_t)Mq * QKVO;

  __bf16* wqkv = (__bf16*)d_ws;
  __bf16* wd   = wqkv + nWqkv;
  __bf16* xbf  = wd + nWd;
  __bf16* qbf  = xbf + nX;
  __bf16* kbf  = qbf + nQ;
  __bf16* vbf  = kbf + nKV;
  __bf16* ctx  = vbf + nKV;
  float*  fused = (float*)(ctx + nCtx);

  const int T = 256;
  const size_t CV = (size_t)T * 8;   // elements per block in f2bf
  f2bf_kernel<<<(unsigned)((nWqkv + CV - 1) / CV), T, 0, stream>>>(W_qkv, wqkv, nWqkv);
  f2bf_kernel<<<(unsigned)((nWd   + CV - 1) / CV), T, 0, stream>>>(W_dense, wd, nWd);
  f2bf_kernel<<<(unsigned)((nX    + CV - 1) / CV), T, 0, stream>>>(query, xbf, nX);

  // fused = query @ W_qkv + b_qkv   (2048 x 4672, K = 4544)
  gemm_bf16_kernel<<<dim3(QKVO / 64, Mq / 64), 128, 0, stream>>>(
      xbf, wqkv, b_qkv, fused, Mq, QKVO, Hq);

  rope_split_kernel<<<(unsigned)((nFused + T - 1) / T), T, 0, stream>>>(
      fused, qbf, kbf, vbf);

  mqa_flash_kernel<<<dim3(Bq * NHq, Sq / 64), 128, 0, stream>>>(
      qbf, kbf, vbf, ctx);

  // out = ctx @ W_dense + b_dense   (2048 x 4544, K = 4544)
  gemm_bf16_kernel<<<dim3(Hq / 64, Mq / 64), 128, 0, stream>>>(
      ctx, wd, b_dense, out, Mq, Hq, Hq);
}
